// PPGNN_20289425506401
// MI455X (gfx1250) — matrix operand order repaired and checked by reading, not compiled, not run on hardware
//
#include <hip/hip_runtime.h>
#include <cstddef>
#include <cstdint>

// ---------------------------------------------------------------------------
// PPGNN for MI455X (gfx1250, wave32).
//   * 30 sparse aggregations dominate; N*128 floats = 51.2MB -> L2-resident.
//   * CSR built once per launch -> pull-style gather, no atomics in hot loop.
//   * Dense GEMMs (input lift + readout) use v_wmma_f32_16x16x32_f16.
// ---------------------------------------------------------------------------

typedef __attribute__((ext_vector_type(16))) _Float16 v16h;
typedef __attribute__((ext_vector_type(8)))  float    v8f;

constexpr int   NN   = 100000;
constexpr int   EE   = 1600000;
constexpr int   INCH = 128;
constexpr int   HIDC = 64;
constexpr int   NCLS = 40;
constexpr int   LAYERS = 15;
constexpr float DTC  = 0.1f;

// ------------------------------- utilities --------------------------------

__global__ void ppgnn_zero_i32(int* __restrict__ p, int n) {
    int i = blockIdx.x * blockDim.x + threadIdx.x;
    if (i < n) p[i] = 0;
}

__global__ void ppgnn_count_deg(const int* __restrict__ dst, int* __restrict__ deg) {
    int e = blockIdx.x * blockDim.x + threadIdx.x;
    if (e < EE) atomicAdd(&deg[dst[e]], 1);
}

__global__ void ppgnn_dinv(const int* __restrict__ deg, float* __restrict__ dinv) {
    int i = blockIdx.x * blockDim.x + threadIdx.x;
    if (i < NN) {
        float d = (float)deg[i];
        dinv[i] = rsqrtf(fmaxf(d, 1.0f));
    }
}

// Single-block exclusive prefix scan over deg[0..NN) -> row_ptr[0..NN].
__global__ void ppgnn_scan(const int* __restrict__ deg, int* __restrict__ row_ptr) {
    __shared__ int partial[1024];
    const int T = 1024;
    const int t = threadIdx.x;
    const int chunk = (NN + T - 1) / T;
    const int beg = t * chunk;
    const int end = min(beg + chunk, NN);
    int s = 0;
    for (int i = beg; i < end; ++i) s += deg[i];
    partial[t] = s;
    __syncthreads();
    if (t == 0) {
        int run = 0;
        for (int i = 0; i < T; ++i) { int v = partial[i]; partial[i] = run; run += v; }
    }
    __syncthreads();
    int run = partial[t];
    for (int i = beg; i < end; ++i) { row_ptr[i] = run; run += deg[i]; }
    if (t == T - 1) row_ptr[NN] = run;   // thread 1023's range is empty -> run == E
}

// Slot assignment via atomics (within-row edge order is arbitrary; sums are
// order-independent up to fp rounding).
__global__ void ppgnn_scatter(const int* __restrict__ src, const int* __restrict__ dst,
                              const float* __restrict__ dinv, const int* __restrict__ row_ptr,
                              int* __restrict__ cursor, int* __restrict__ csr_src,
                              float* __restrict__ csr_w) {
    int e = blockIdx.x * blockDim.x + threadIdx.x;
    if (e >= EE) return;
    int s = src[e], d = dst[e];
    int pos = row_ptr[d] + atomicAdd(&cursor[d], 1);
    csr_src[pos] = s;
    csr_w[pos]   = dinv[s] * dinv[d];
}

// ------------------------- input lift GEMM (WMMA) --------------------------
// h[:, :64] = tanh(x @ W_lx + b_lx);  h[:, 64:] = 1.
// Block = 128 threads = 4 waves, 64 rows. LDS-staged f16, 16 WMMAs/wave.
__global__ __launch_bounds__(128)
void ppgnn_in_gemm(const float* __restrict__ x, const float* __restrict__ Wlx,
                   const float* __restrict__ blx, float* __restrict__ h) {
    __shared__ _Float16 xs[64][INCH];   // A tile, row-major
    __shared__ _Float16 wt[64][INCH];   // B transposed: wt[n][k] = W[k][n]
    const int tid  = threadIdx.x;
    const int row0 = blockIdx.x * 64;

    for (int i = tid; i < HIDC * INCH; i += 128) {       // W: 128x64 row-major
        int k = i >> 6, n = i & 63;
        wt[n][k] = (_Float16)Wlx[i];
    }
    for (int i = tid; i < 64 * INCH; i += 128) {
        int r = i >> 7, c = i & 127;
        int g = row0 + r;
        xs[r][c] = (g < NN) ? (_Float16)x[(size_t)g * INCH + c] : (_Float16)0.0f;
    }
    __syncthreads();

    const int wave = tid >> 5, lane = tid & 31;
    const int rloc = wave * 16 + (lane & 15);   // A: lane -> matrix row M
    const int kbA  = (lane >> 4) * 8;           // A: lane-half -> K sub-block
    const int kbB  = (lane >> 4) * 8;           // B: lane-half -> K sub-block

    for (int nt = 0; nt < 4; ++nt) {
        v8f acc = {};
        const int ncol = nt * 16 + (lane & 15); // B: lane -> matrix col N
        for (int ks = 0; ks < 4; ++ks) {
            const int k0 = ks * 32;
            v16h a, b;
#pragma unroll
            for (int j = 0; j < 8; ++j) {       // contiguous -> ds_load_b128-able
                a[j]     = xs[rloc][k0 + kbA + j];
                a[8 + j] = xs[rloc][k0 + 16 + kbA + j];
            }
#pragma unroll
            for (int j = 0; j < 8; ++j) {
                b[j]     = wt[ncol][k0 + kbB + j];
                b[8 + j] = wt[ncol][k0 + 16 + kbB + j];
            }
            acc = __builtin_amdgcn_wmma_f32_16x16x32_f16(false, a, false, b,
                                                         (short)0, acc, false, false);
        }
        const float bias = blx[ncol];
#pragma unroll
        for (int r = 0; r < 8; ++r) {
            int grow = row0 + wave * 16 + (lane >> 4) * 8 + r;   // C/D layout
            if (grow < NN) {
                h[(size_t)grow * INCH + ncol]        = tanhf(acc[r] + bias);
                h[(size_t)grow * INCH + 64 + ncol]   = 1.0f;
            }
        }
    }
}

// ----------------------------- per-layer math ------------------------------

__global__ void ppgnn_rhs(const float* __restrict__ h, float* __restrict__ rhs) {
    int i = blockIdx.x * blockDim.x + threadIdx.x;      // over N*64 (X,Y) pairs
    if (i >= NN * HIDC) return;
    int n = i >> 6, j = i & 63;
    size_t base = (size_t)n * INCH;
    float X = h[base + j];
    float Y = h[base + 64 + j];
    rhs[base + j]      = X + DTC * (X * (1.0f - Y));
    rhs[base + 64 + j] = Y + DTC * (Y * (X - 1.0f));
}

// Jacobi step 1: Z1 = (rhs + DT*agg(rhs)) / (1+DT).
// One wave32 per node; each lane owns a float4 slice of the 128-wide row.
__global__ __launch_bounds__(256)
void ppgnn_jacobi1(const float* __restrict__ rhs, const int* __restrict__ row_ptr,
                   const int* __restrict__ csr_src, const float* __restrict__ csr_w,
                   float* __restrict__ Z1) {
    const int wave = threadIdx.x >> 5;
    const int lane = threadIdx.x & 31;
    const int node = blockIdx.x * 8 + wave;
    if (node >= NN) return;
    const int beg = row_ptr[node], end = row_ptr[node + 1];
    float4 a0 = {0, 0, 0, 0}, a1 = {0, 0, 0, 0};
    int p = beg;
    for (; p + 1 < end; p += 2) {                         // 2x unroll for ILP
        int   s0 = csr_src[p],     s1 = csr_src[p + 1];
        float w0 = csr_w[p],       w1 = csr_w[p + 1];
        float4 z0 = reinterpret_cast<const float4*>(rhs + (size_t)s0 * INCH)[lane];
        float4 z1 = reinterpret_cast<const float4*>(rhs + (size_t)s1 * INCH)[lane];
        a0.x += w0 * z0.x; a0.y += w0 * z0.y; a0.z += w0 * z0.z; a0.w += w0 * z0.w;
        a1.x += w1 * z1.x; a1.y += w1 * z1.y; a1.z += w1 * z1.z; a1.w += w1 * z1.w;
    }
    if (p < end) {
        int s = csr_src[p]; float w = csr_w[p];
        float4 z = reinterpret_cast<const float4*>(rhs + (size_t)s * INCH)[lane];
        a0.x += w * z.x; a0.y += w * z.y; a0.z += w * z.z; a0.w += w * z.w;
    }
    a0.x += a1.x; a0.y += a1.y; a0.z += a1.z; a0.w += a1.w;
    const float4 r = reinterpret_cast<const float4*>(rhs + (size_t)node * INCH)[lane];
    const float inv = 1.0f / (1.0f + DTC);
    float4 z;
    z.x = (r.x + DTC * a0.x) * inv; z.y = (r.y + DTC * a0.y) * inv;
    z.z = (r.z + DTC * a0.z) * inv; z.w = (r.w + DTC * a0.w) * inv;
    reinterpret_cast<float4*>(Z1 + (size_t)node * INCH)[lane] = z;
}

// Jacobi step 2 fused with the tau-blend update (h updated in place: each
// node only reads/writes its own row of h).
__global__ __launch_bounds__(256)
void ppgnn_jacobi2_update(const float* __restrict__ rhs, const float* __restrict__ Z1,
                          const int* __restrict__ row_ptr, const int* __restrict__ csr_src,
                          const float* __restrict__ csr_w, const float* __restrict__ taus,
                          int layer, float* __restrict__ h) {
    const int wave = threadIdx.x >> 5;
    const int lane = threadIdx.x & 31;
    const int node = blockIdx.x * 8 + wave;
    if (node >= NN) return;
    const int beg = row_ptr[node], end = row_ptr[node + 1];
    float4 a0 = {0, 0, 0, 0}, a1 = {0, 0, 0, 0};
    int p = beg;
    for (; p + 1 < end; p += 2) {
        int   s0 = csr_src[p],     s1 = csr_src[p + 1];
        float w0 = csr_w[p],       w1 = csr_w[p + 1];
        float4 z0 = reinterpret_cast<const float4*>(Z1 + (size_t)s0 * INCH)[lane];
        float4 z1 = reinterpret_cast<const float4*>(Z1 + (size_t)s1 * INCH)[lane];
        a0.x += w0 * z0.x; a0.y += w0 * z0.y; a0.z += w0 * z0.z; a0.w += w0 * z0.w;
        a1.x += w1 * z1.x; a1.y += w1 * z1.y; a1.z += w1 * z1.z; a1.w += w1 * z1.w;
    }
    if (p < end) {
        int s = csr_src[p]; float w = csr_w[p];
        float4 z = reinterpret_cast<const float4*>(Z1 + (size_t)s * INCH)[lane];
        a0.x += w * z.x; a0.y += w * z.y; a0.z += w * z.z; a0.w += w * z.w;
    }
    a0.x += a1.x; a0.y += a1.y; a0.z += a1.z; a0.w += a1.w;
    const float4 r  = reinterpret_cast<const float4*>(rhs + (size_t)node * INCH)[lane];
    float4 hv = reinterpret_cast<const float4*>(h + (size_t)node * INCH)[lane];
    const float inv = 1.0f / (1.0f + DTC);
    const float tau = 1.0f / (1.0f + expf(-taus[layer]));
    const float omt = 1.0f - tau;
    hv.x = omt * hv.x + tau * ((r.x + DTC * a0.x) * inv);
    hv.y = omt * hv.y + tau * ((r.y + DTC * a0.y) * inv);
    hv.z = omt * hv.z + tau * ((r.z + DTC * a0.z) * inv);
    hv.w = omt * hv.w + tau * ((r.w + DTC * a0.w) * inv);
    reinterpret_cast<float4*>(h + (size_t)node * INCH)[lane] = hv;
}

// ----------------------------- readout GEMM --------------------------------
// out = logit_scale * h[:, :64] @ W_ro + b_ro   (40 cols padded to 48)
__global__ __launch_bounds__(128)
void ppgnn_readout(const float* __restrict__ h, const float* __restrict__ Wro,
                   const float* __restrict__ bro, const float* __restrict__ scale_p,
                   float* __restrict__ out) {
    __shared__ _Float16 hs[64][64];
    __shared__ _Float16 wt[48][64];     // wt[c][k] = W_ro[k][c], zero-padded
    const int tid  = threadIdx.x;
    const int row0 = blockIdx.x * 64;

    for (int i = tid; i < 64 * 64; i += 128) {
        int r = i >> 6, c = i & 63;
        int g = row0 + r;
        hs[r][c] = (g < NN) ? (_Float16)h[(size_t)g * INCH + c] : (_Float16)0.0f;
    }
    for (int i = tid; i < 48 * 64; i += 128) {
        int c = i >> 6, k = i & 63;
        wt[c][k] = (c < NCLS) ? (_Float16)Wro[k * NCLS + c] : (_Float16)0.0f;
    }
    __syncthreads();

    const float scale = scale_p[0];
    const int wave = tid >> 5, lane = tid & 31;
    const int rloc = wave * 16 + (lane & 15);
    const int kbA  = (lane >> 4) * 8;
    const int kbB  = (lane >> 4) * 8;

    for (int nt = 0; nt < 3; ++nt) {
        v8f acc = {};
        const int ncol = nt * 16 + (lane & 15);
        for (int ks = 0; ks < 2; ++ks) {
            const int k0 = ks * 32;
            v16h a, b;
#pragma unroll
            for (int j = 0; j < 8; ++j) {
                a[j]     = hs[rloc][k0 + kbA + j];
                a[8 + j] = hs[rloc][k0 + 16 + kbA + j];
            }
#pragma unroll
            for (int j = 0; j < 8; ++j) {
                b[j]     = wt[ncol][k0 + kbB + j];
                b[8 + j] = wt[ncol][k0 + 16 + kbB + j];
            }
            acc = __builtin_amdgcn_wmma_f32_16x16x32_f16(false, a, false, b,
                                                         (short)0, acc, false, false);
        }
        if (ncol < NCLS) {
            const float bias = bro[ncol];
#pragma unroll
            for (int r = 0; r < 8; ++r) {
                int grow = row0 + wave * 16 + (lane >> 4) * 8 + r;
                if (grow < NN) out[(size_t)grow * NCLS + ncol] = scale * acc[r] + bias;
            }
        }
    }
}

// ------------------------------- launcher ----------------------------------

extern "C" void kernel_launch(void* const* d_in, const int* in_sizes, int n_in,
                              void* d_out, int out_size, void* d_ws, size_t ws_size,
                              hipStream_t stream) {
    (void)in_sizes; (void)n_in; (void)out_size; (void)ws_size;

    const float* x     = (const float*)d_in[0];
    const float* Wlx   = (const float*)d_in[1];
    const float* blx   = (const float*)d_in[2];
    // d_in[3], d_in[4] (W_ly, b_ly) are unused by the reference.
    const float* taus  = (const float*)d_in[5];
    const float* lscal = (const float*)d_in[6];
    const float* Wro   = (const float*)d_in[7];
    const float* bro   = (const float*)d_in[8];
    const int*   edges = (const int*)d_in[9];
    const int*   esrc  = edges;
    const int*   edst  = edges + EE;
    float*       out   = (float*)d_out;

    // Workspace carve-up (~168 MB total), 256B-aligned slices.
    char*  ws  = (char*)d_ws;
    size_t off = 0;
    auto carve = [&](size_t bytes) -> void* {
        void* p = ws + off;
        off += (bytes + 255) & ~(size_t)255;
        return p;
    };
    float* h       = (float*)carve((size_t)NN * INCH * 4);
    float* rhs     = (float*)carve((size_t)NN * INCH * 4);
    float* Z1      = (float*)carve((size_t)NN * INCH * 4);
    int*   degcur  = (int*)  carve((size_t)2 * NN * 4);    // deg | cursor
    int*   deg     = degcur;
    int*   cursor  = degcur + NN;
    float* dinv    = (float*)carve((size_t)NN * 4);
    int*   row_ptr = (int*)  carve((size_t)(NN + 1) * 4);
    int*   csr_src = (int*)  carve((size_t)EE * 4);
    float* csr_w   = (float*)carve((size_t)EE * 4);

    const int T = 256;
    const int gE    = (EE + T - 1) / T;
    const int gN    = (NN + T - 1) / T;
    const int g2N   = (2 * NN + T - 1) / T;
    const int gElem = (NN * HIDC + T - 1) / T;
    const int gNode = (NN + 7) / 8;          // 8 waves/block, 1 node/wave
    const int gGemm = (NN + 63) / 64;        // 64 rows/block

    // ---- CSR build (per launch; counters re-zeroed for graph replay) ----
    ppgnn_zero_i32<<<g2N, T, 0, stream>>>(degcur, 2 * NN);
    ppgnn_count_deg<<<gE, T, 0, stream>>>(edst, deg);
    ppgnn_dinv<<<gN, T, 0, stream>>>(deg, dinv);
    ppgnn_scan<<<1, 1024, 0, stream>>>(deg, row_ptr);
    ppgnn_scatter<<<gE, T, 0, stream>>>(esrc, edst, dinv, row_ptr, cursor,
                                        csr_src, csr_w);

    // ---- input lift (WMMA) ----
    ppgnn_in_gemm<<<gGemm, 128, 0, stream>>>(x, Wlx, blx, h);

    // ---- 15 layers x (rhs, jacobi1, jacobi2+update) ----
    for (int l = 0; l < LAYERS; ++l) {
        ppgnn_rhs<<<gElem, T, 0, stream>>>(h, rhs);
        ppgnn_jacobi1<<<gNode, T, 0, stream>>>(rhs, row_ptr, csr_src, csr_w, Z1);
        ppgnn_jacobi2_update<<<gNode, T, 0, stream>>>(rhs, Z1, row_ptr, csr_src,
                                                      csr_w, taus, l, h);
    }

    // ---- readout (WMMA) ----
    ppgnn_readout<<<gGemm, 128, 0, stream>>>(h, Wro, bro, lscal, out);
}
